// SparseAttentionChunkLayer_7834020348441
// MI455X (gfx1250) — matrix operand order repaired and checked by reading, not compiled, not run on hardware
//
#include <hip/hip_runtime.h>
#include <cmath>

// ---------------------------------------------------------------------------
// Types
// ---------------------------------------------------------------------------
typedef __attribute__((ext_vector_type(16))) __bf16          bf16x16;
typedef __attribute__((ext_vector_type(8)))  float           f32x8;
typedef __attribute__((ext_vector_type(8)))  unsigned short  u16x8;
typedef __attribute__((ext_vector_type(4)))  unsigned int    u32x4;
typedef __attribute__((ext_vector_type(8)))  int             i32x8;
typedef __attribute__((ext_vector_type(4)))  int             i32x4;

// therock/amdgpu-toolchain ships the TDM header and uses the 6-arg builtin;
// ROCm 7.2 (clang-22) uses the 5-arg form.
#if __has_include(<hip/amd_detail/amd_gfx1250_TDM.h>)
#define ATHENA_TDM_6ARG 1
#endif

__device__ __forceinline__ unsigned short f2bf_bits(float f) {
    unsigned u = __builtin_bit_cast(unsigned, f);
    unsigned r = u + 0x7FFFu + ((u >> 16) & 1u);   // round-to-nearest-even
    return (unsigned short)(r >> 16);
}

__device__ __forceinline__ unsigned lds_offset_of(const void* p) {
    // AS(3) pointers on amdgcn are 32-bit LDS byte offsets.
    return (unsigned)(unsigned long long)
           (__attribute__((address_space(3))) const void*)p;
}

// ---------------------------------------------------------------------------
// TDM: 2D tile load Global -> LDS (Tensor DMA Descriptor per ISA ch.8).
//   tile_d0 = contiguous elements per line, tile_d1 = #lines,
//   stride  = elements between consecutive lines, data_size = 2 bytes (bf16).
// Issued per-wave (EXEC ignored); caller gates to a single wave.
// ---------------------------------------------------------------------------
__device__ __forceinline__ void tdm_load_2d(const void* gaddr, unsigned lds_off,
                                            unsigned tile_d0, unsigned tile_d1,
                                            unsigned long long stride_elems)
{
    const unsigned long long ga = (unsigned long long)gaddr;
    u32x4 g0;
    g0[0] = 1u;                                     // count=1, user mode
    g0[1] = lds_off;                                // lds_addr (bytes)
    g0[2] = (unsigned)ga;                           // global_addr[31:0]
    g0[3] = (unsigned)((ga >> 32) & 0x1FFFFFFu)     // global_addr[56:32]
          | (2u << 30);                             // type = 2 ("image")
    i32x8 g1;
    g1[0] = (int)(1u << 16);                        // data_size = 1 -> 2 bytes
    g1[1] = (int)((tile_d0 & 0xFFFFu) << 16);       // tensor_dim0[15:0]
    g1[2] = (int)((tile_d0 >> 16) | ((tile_d1 & 0xFFFFu) << 16)); // td0 hi|td1 lo
    g1[3] = (int)((tile_d1 >> 16) | ((tile_d0 & 0xFFFFu) << 16)); // td1 hi|tile_dim0
    g1[4] = (int)(tile_d1 & 0xFFFFu);               // tile_dim1 (tile_dim2 = 0)
    g1[5] = (int)(unsigned)(stride_elems & 0xFFFFFFFFull);        // dim0_stride lo
    g1[6] = (int)(unsigned)((stride_elems >> 32) & 0xFFFFull);    // dim0_stride hi
    g1[7] = 0;                                      // dim1_stride = 0 (2D tile)
    i32x4 z4 = {0, 0, 0, 0};
#ifdef ATHENA_TDM_6ARG
    i32x8 z8 = {0, 0, 0, 0, 0, 0, 0, 0};
    __builtin_amdgcn_tensor_load_to_lds(g0, g1, z4, z4, z8, 0);
#else
    __builtin_amdgcn_tensor_load_to_lds(g0, g1, z4, z4, 0);
#endif
}

// ---------------------------------------------------------------------------
// f32 -> bf16 cast with per-batch row remap (cached K/V land inside the
// full-K buffer at rows [0, Kc) of each batch).
// ---------------------------------------------------------------------------
__global__ __launch_bounds__(256) void cast_f32_bf16_rows(
    const float* __restrict__ src, unsigned short* __restrict__ dst,
    long long total, long long elemsPerBatch, long long dstBatchStride)
{
    for (long long i = (long long)blockIdx.x * blockDim.x + threadIdx.x;
         i < total; i += (long long)gridDim.x * blockDim.x) {
        long long b   = i / elemsPerBatch;
        long long rem = i - b * elemsPerBatch;
        dst[b * dstBatchStride + rem] = f2bf_bits(src[i]);
    }
}

// ---------------------------------------------------------------------------
// bf16 transpose  dst[z][c][r] = src[z][r][c]
// ---------------------------------------------------------------------------
__global__ __launch_bounds__(256) void transpose_bf16(
    const unsigned short* __restrict__ src, unsigned short* __restrict__ dst,
    int rows, int cols)
{
    __shared__ unsigned short t[32][33];
    const long long z = blockIdx.z;
    const unsigned short* s = src + z * (long long)rows * cols;
    unsigned short*       d = dst + z * (long long)rows * cols;
    const int r0 = blockIdx.x * 32;
    const int c0 = blockIdx.y * 32;
    const int tx = threadIdx.x, ty = threadIdx.y;        // 32 x 8
#pragma unroll
    for (int i = 0; i < 32; i += 8)
        t[ty + i][tx] = s[(long long)(r0 + ty + i) * cols + (c0 + tx)];
    __syncthreads();
#pragma unroll
    for (int i = 0; i < 32; i += 8)
        d[(long long)(c0 + ty + i) * rows + (r0 + tx)] = t[tx][ty + i];
}

// ---------------------------------------------------------------------------
// Generic NT GEMM, bf16 in / f32 accum via V_WMMA_F32_16X16X32_BF16.
//   C[z][m][n] = alpha * sum_k A[z][m][k]*B[z][n][k] + bias[n]  (+ causal mask)
// Block = 256 threads = 8 waves; tile BM=128, BN=64, BK=32, double-buffered
// LDS fed by the Tensor Data Mover (wave 0 issues, TENSORcnt pipelined).
// All launched shapes divide the tile sizes exactly (no edge guards needed).
// ---------------------------------------------------------------------------
__global__ __launch_bounds__(256) void gemm_nt_bf16(
    const unsigned short* __restrict__ A, int lda, long long strideA,
    const unsigned short* __restrict__ Bm, int ldb, long long strideB,
    float* __restrict__ Cf, unsigned short* __restrict__ Ch,
    int ldc, long long strideC,
    const float* __restrict__ bias,
    int K, float alpha, int maskKc)
{
    __shared__ __align__(16) unsigned short As[2][128 * 32];
    __shared__ __align__(16) unsigned short Bs[2][64 * 32];

    const int tid  = threadIdx.x;
    const int wid  = tid >> 5;
    const int lane = tid & 31;
    const int z  = blockIdx.z;
    const int mb = blockIdx.y;
    const int nb = blockIdx.x;

    const unsigned short* Ab = A  + (long long)z * strideA + (long long)(mb * 128) * lda;
    const unsigned short* Bb = Bm + (long long)z * strideB + (long long)(nb *  64) * ldb;

    const unsigned offA[2] = { lds_offset_of(&As[0][0]), lds_offset_of(&As[1][0]) };
    const unsigned offB[2] = { lds_offset_of(&Bs[0][0]), lds_offset_of(&Bs[1][0]) };

    f32x8 acc[4];
#pragma unroll
    for (int i = 0; i < 4; ++i) acc[i] = f32x8{0.f,0.f,0.f,0.f,0.f,0.f,0.f,0.f};

    // fragment lane geometry (per ISA VGPR layout tables)
    const int l15 = lane & 15;
    const int akb = (lane >> 4) * 8;     // A: K{0-7,16-23} vs K{8-15,24-31}
    const int bkb = (lane >> 4) * 8;     // B: K{0-7,16-23} vs K{8-15,24-31}
    const int aoff = (wid * 16 + l15) * 32 + akb;

    const int nk = K / 32;
    // prologue: stage tiles for step 0 into buffer 0
    if (wid == 0) {
        tdm_load_2d(Ab, offA[0], 32u, 128u, (unsigned long long)lda);
        tdm_load_2d(Bb, offB[0], 32u,  64u, (unsigned long long)ldb);
    }

    for (int t = 0; t < nk; ++t) {
        const int cur = t & 1;
        if (wid == 0) {
            if (t + 1 < nk) {   // stage step t+1 into the other buffer
                tdm_load_2d(Ab + (long long)(t + 1) * 32, offA[cur ^ 1],
                            32u, 128u, (unsigned long long)lda);
                tdm_load_2d(Bb + (long long)(t + 1) * 32, offB[cur ^ 1],
                            32u, 64u, (unsigned long long)ldb);
                __builtin_amdgcn_s_wait_tensorcnt((short)2);  // step-t tiles done
            } else {
                __builtin_amdgcn_s_wait_tensorcnt((short)0);
            }
        }
        __syncthreads();   // buffer `cur` visible to all waves

        const unsigned short* Asb = As[cur];
        const unsigned short* Bsb = Bs[cur];

        u16x8 alo = *(const u16x8*)(Asb + aoff);
        u16x8 ahi = *(const u16x8*)(Asb + aoff + 16);
        bf16x16 afrag = __builtin_bit_cast(bf16x16,
            __builtin_shufflevector(alo, ahi, 0,1,2,3,4,5,6,7,8,9,10,11,12,13,14,15));

        bf16x16 bfrag[4];
#pragma unroll
        for (int nt = 0; nt < 4; ++nt) {   // gather all B fragments first
            const int boff = (nt * 16 + l15) * 32 + bkb;
            u16x8 blo = *(const u16x8*)(Bsb + boff);
            u16x8 bhi = *(const u16x8*)(Bsb + boff + 16);
            bfrag[nt] = __builtin_bit_cast(bf16x16,
                __builtin_shufflevector(blo, bhi, 0,1,2,3,4,5,6,7,8,9,10,11,12,13,14,15));
        }
#pragma unroll
        for (int nt = 0; nt < 4; ++nt)     // then stream 4 back-to-back WMMAs
            acc[nt] = __builtin_amdgcn_wmma_f32_16x16x32_bf16(
                false, afrag, false, bfrag[nt], (short)0, acc[nt], false, false);

        __syncthreads();   // all waves done with buffer `cur` before reuse
    }

    // epilogue: C layout — VGPR i: M=i (lanes 0-15) / M=8+i (lanes 16-31)
    const int mrow0 = mb * 128 + wid * 16 + akb;   // akb == (lane<16 ? 0 : 8)
#pragma unroll
    for (int nt = 0; nt < 4; ++nt) {
        const int n  = nb * 64 + nt * 16 + l15;
        const float bv = bias ? bias[n] : 0.0f;
#pragma unroll
        for (int i = 0; i < 8; ++i) {
            const int m = mrow0 + i;
            float v = acc[nt][i] * alpha + bv;
            if (maskKc >= 0 && n >= maskKc && (n - maskKc) > m)
                v = -__builtin_inff();
            const long long idx = (long long)z * strideC + (long long)m * ldc + n;
            if (Cf) Cf[idx] = v;
            if (Ch) Ch[idx] = f2bf_bits(v);
        }
    }
}

// ---------------------------------------------------------------------------
// Row softmax over S[row][0..Kn), then write bf16 probabilities IN PLACE over
// the row's f32 storage (row cached in LDS first, so aliasing is safe).
// grid = (L, B), block = 256.
// ---------------------------------------------------------------------------
#define SMAX_ROW 5120
__global__ __launch_bounds__(256) void softmax_row_bf16(
    float* __restrict__ S, int Kn)
{
    __shared__ float buf[SMAX_ROW];
    __shared__ float red[256];
    const int tid = threadIdx.x;
    const long long row = (long long)blockIdx.y * gridDim.x + blockIdx.x;
    float* rp = S + row * (long long)Kn;

    float lmax = -__builtin_inff();
    for (int k = tid; k < Kn; k += 256) {
        float v = rp[k];
        buf[k] = v;
        lmax = fmaxf(lmax, v);
    }
    red[tid] = lmax;
    __syncthreads();
    for (int s = 128; s > 0; s >>= 1) {
        if (tid < s) red[tid] = fmaxf(red[tid], red[tid + s]);
        __syncthreads();
    }
    const float mx = red[0];
    __syncthreads();

    float lsum = 0.f;
    for (int k = tid; k < Kn; k += 256) lsum += __expf(buf[k] - mx);
    red[tid] = lsum;
    __syncthreads();
    for (int s = 128; s > 0; s >>= 1) {
        if (tid < s) red[tid] += red[tid + s];
        __syncthreads();
    }
    const float inv = 1.0f / red[0];
    __syncthreads();

    unsigned short* op = (unsigned short*)rp;   // bf16 row over same storage
    for (int k = tid; k < Kn; k += 256)
        op[k] = f2bf_bits(__expf(buf[k] - mx) * inv);
}

// ---------------------------------------------------------------------------
// Host orchestration
// ---------------------------------------------------------------------------
extern "C" void kernel_launch(void* const* d_in, const int* in_sizes, int n_in,
                              void* d_out, int out_size, void* d_ws, size_t ws_size,
                              hipStream_t stream)
{
    const float* chunk    = (const float*)d_in[0];
    const float* cached_k = (const float*)d_in[1];
    const float* cached_v = (const float*)d_in[2];
    const float* Wq = (const float*)d_in[5];  const float* bq = (const float*)d_in[6];
    const float* Wk = (const float*)d_in[7];  const float* bk = (const float*)d_in[8];
    const float* Wv = (const float*)d_in[9];  const float* bv = (const float*)d_in[10];
    const float* Wo = (const float*)d_in[11]; const float* bo = (const float*)d_in[12];
    float* out = (float*)d_out;

    const int  Bn = 4, L = 4096, D = 2048;
    const int  Kc = in_sizes[3];          // 576
    const int  KN = Kc + L;               // 4672
    const float scale = 1.0f / sqrtf((float)D);

    // workspace carve-up (bf16 stored as unsigned short)
    char* ws = (char*)d_ws;
    size_t off = 0;
    auto alloc = [&](size_t bytes) -> void* {
        void* p = ws + off;
        off += (bytes + 255) & ~(size_t)255;
        return p;
    };
    const long long LD  = (long long)L * D;       // per-batch activation elems
    const long long KND = (long long)KN * D;      // per-batch full-K/V elems
    unsigned short* Wqb = (unsigned short*)alloc((size_t)D * D * 2);
    unsigned short* Wkb = (unsigned short*)alloc((size_t)D * D * 2);
    unsigned short* Wvb = (unsigned short*)alloc((size_t)D * D * 2);
    unsigned short* Wob = (unsigned short*)alloc((size_t)D * D * 2);
    unsigned short* Xb  = (unsigned short*)alloc((size_t)Bn * LD * 2);
    unsigned short* Qb  = (unsigned short*)alloc((size_t)Bn * LD * 2);
    unsigned short* Kf  = (unsigned short*)alloc((size_t)Bn * KND * 2);
    unsigned short* Vf  = (unsigned short*)alloc((size_t)Bn * KND * 2);
    unsigned short* Vt  = (unsigned short*)alloc((size_t)Bn * KND * 2);
    float*          Sbuf= (float*)alloc((size_t)Bn * L * KN * 4);
    unsigned short* Ab  = (unsigned short*)alloc((size_t)Bn * LD * 2);

    const dim3 blk(256);
    const dim3 cgrid(4096);

    // 1) casts: weights + chunk (flat), cached K/V into full buffers rows [0,Kc)
    {
        const long long wTot = (long long)D * D;
        cast_f32_bf16_rows<<<cgrid, blk, 0, stream>>>(Wq, Wqb, wTot, wTot, wTot);
        cast_f32_bf16_rows<<<cgrid, blk, 0, stream>>>(Wk, Wkb, wTot, wTot, wTot);
        cast_f32_bf16_rows<<<cgrid, blk, 0, stream>>>(Wv, Wvb, wTot, wTot, wTot);
        cast_f32_bf16_rows<<<cgrid, blk, 0, stream>>>(Wo, Wob, wTot, wTot, wTot);
        cast_f32_bf16_rows<<<cgrid, blk, 0, stream>>>(chunk, Xb, Bn * LD, Bn * LD, Bn * LD);
        const long long cTot = (long long)Bn * Kc * D;
        cast_f32_bf16_rows<<<cgrid, blk, 0, stream>>>(cached_k, Kf, cTot, (long long)Kc * D, KND);
        cast_f32_bf16_rows<<<cgrid, blk, 0, stream>>>(cached_v, Vf, cTot, (long long)Kc * D, KND);
    }

    // 2) projections: X[z] (L x D) @ W^T -> bf16
    {
        dim3 g(D / 64, L / 128, Bn);
        gemm_nt_bf16<<<g, blk, 0, stream>>>(Xb, D, LD, Wqb, D, 0,
                                            nullptr, Qb, D, LD, bq, D, 1.0f, -1);
        gemm_nt_bf16<<<g, blk, 0, stream>>>(Xb, D, LD, Wkb, D, 0,
                                            nullptr, Kf + (long long)Kc * D, D, KND,
                                            bk, D, 1.0f, -1);
        gemm_nt_bf16<<<g, blk, 0, stream>>>(Xb, D, LD, Wvb, D, 0,
                                            nullptr, Vf + (long long)Kc * D, D, KND,
                                            bv, D, 1.0f, -1);
    }

    // 3) Vt[z][d][k] = Vf[z][k][d]
    {
        dim3 g(KN / 32, D / 32, Bn), b2(32, 8);
        transpose_bf16<<<g, b2, 0, stream>>>(Vf, Vt, KN, D);
    }

    // 4) S = scale * Q @ Kf^T, causal mask on new keys -> f32
    {
        dim3 g(KN / 64, L / 128, Bn);
        gemm_nt_bf16<<<g, blk, 0, stream>>>(Qb, D, LD, Kf, D, KND,
                                            Sbuf, nullptr, KN, (long long)L * KN,
                                            nullptr, D, scale, Kc);
    }

    // 5) softmax rows, bf16 P written in place (row stride stays KN f32 slots)
    softmax_row_bf16<<<dim3(L, Bn), blk, 0, stream>>>(Sbuf, KN);

    // 6) attended = P @ Vt^T -> bf16   (P: bf16 elems, lda counted in ushort = 2*KN)
    {
        dim3 g(D / 64, L / 128, Bn);
        gemm_nt_bf16<<<g, blk, 0, stream>>>((const unsigned short*)Sbuf, 2 * KN,
                                            (long long)L * 2 * KN,
                                            Vt, KN, KND,
                                            nullptr, Ab, D, LD,
                                            nullptr, KN, 1.0f, -1);
    }

    // 7) out = attended @ Wo^T + bo -> f32
    {
        dim3 g(D / 64, L / 128, Bn);
        gemm_nt_bf16<<<g, blk, 0, stream>>>(Ab, D, LD, Wob, D, 0,
                                            out, nullptr, D, LD, bo, D, 1.0f, -1);
    }
    (void)n_in; (void)out_size; (void)ws_size;
}
